// FlexBertUnpadRopeAttention_11278584119786
// MI455X (gfx1250) — matrix operand order repaired and checked by reading, not compiled, not run on hardware
//
#include <hip/hip_runtime.h>
#include <hip/hip_bf16.h>

typedef __bf16 bf16;
typedef __attribute__((ext_vector_type(16))) __bf16 v16bf;
typedef __attribute__((ext_vector_type(8)))  float  v8f;
typedef __attribute__((ext_vector_type(4)))  unsigned u32x4;
typedef __attribute__((ext_vector_type(8)))  unsigned uv8;
typedef __attribute__((ext_vector_type(8)))  int      i32x8;
typedef __attribute__((ext_vector_type(4)))  int      i32x4;

#define HIDDEN 1024
#define HEADS  16
#define HEAD   64
#define BATCH  8
#define SEQ    1024
#define NTOK   (BATCH * SEQ)
#define QKV_N  (3 * HIDDEN)
#define WIN    64
#define KTILE  208  // V rows staged per block: 64(left)+64(queries)+80(right reach)
#define VROW   66   // V LDS row stride elems (132B = 33 dw, gcd(33,64)=1: conflict-free)
#define KCOL   258  // Kt LDS row stride elems (516B = 129 dw, gcd(129,64)=1)

// ---------- bf16 helpers (round-to-nearest-even) ----------
__device__ inline bf16 f2bf(float f) {
  unsigned u = __builtin_bit_cast(unsigned, f);
  unsigned r = (u + 0x7FFFu + ((u >> 16) & 1u)) >> 16;
  return __builtin_bit_cast(bf16, (unsigned short)r);
}
__device__ inline float bf2f(bf16 b) {
  unsigned u = ((unsigned)__builtin_bit_cast(unsigned short, b)) << 16;
  return __builtin_bit_cast(float, u);
}
__device__ inline bf16 ld_bf(const float* p) { return f2bf(*p); }
__device__ inline bf16 ld_bf(const bf16*  p) { return *p; }
__device__ inline void st_out(bf16*  p, float v) { *p = f2bf(v); }
__device__ inline void st_out(float* p, float v) { *p = v; }

// 16 contiguous bf16 -> fragment, as 8 x b32 loads (p must be 4B-aligned)
__device__ inline v16bf ld16bf(const bf16* p) {
  const unsigned* q = (const unsigned*)p;
  uv8 r;
  #pragma unroll
  for (int i = 0; i < 8; ++i) r[i] = q[i];
  return __builtin_bit_cast(v16bf, r);
}
// two runs of 8 contiguous bf16 (A-fragment halves), as b32 loads
__device__ inline v16bf ld16bf_2x8(const bf16* p0, const bf16* p1) {
  const unsigned* a = (const unsigned*)p0;
  const unsigned* b = (const unsigned*)p1;
  uv8 r;
  #pragma unroll
  for (int i = 0; i < 4; ++i) { r[i] = a[i]; r[4 + i] = b[i]; }
  return __builtin_bit_cast(v16bf, r);
}

// ============================================================
// TDM 2D tile load, D# per CDNA5 ISA §8 (6-arg builtin form).
// tile_d0 elems per row (dim0 contiguous), tile_d1 rows,
// tensor dims for OOB zero-fill, d0stride in elems, LDS row
// padding codes (interval: 2^(c+1) dwords, amount: c+1 dwords).
// ============================================================
__device__ inline void tdm_load_2d(unsigned lds_byte_addr, const bf16* gsrc,
                                   int tile_d0, int tile_d1, int tdim0,
                                   int tdim1, int d0stride, int padi, int pada) {
  unsigned long long ga = (unsigned long long)(size_t)gsrc;
  u32x4 g0;
  g0[0] = 1u;                                   // count=1, user descriptor
  g0[1] = lds_byte_addr;
  g0[2] = (unsigned)(ga & 0xFFFFFFFFu);
  g0[3] = (unsigned)((ga >> 32) & 0x01FFFFFFu) | (2u << 30);  // type=2
  i32x8 g1;
  g1[0] = (1 << 16)        // data_size = 2 bytes
        | (1 << 20)        // pad_enable
        | (padi << 22) | (pada << 25);
  g1[1] = (int)((tdim0 & 0xFFFF) << 16);                     // tensor_dim0 lo16
  g1[2] = (int)((tdim0 >> 16) & 0xFFFF) | ((tdim1 & 0xFFFF) << 16);
  g1[3] = (int)((tdim1 >> 16) & 0xFFFF) | (tile_d0 << 16);   // | tile_dim0
  g1[4] = tile_d1 & 0xFFFF;                                  // tile_dim1 (dim2=0)
  g1[5] = d0stride;                                          // dim0 stride lo32
  g1[6] = 0;
  g1[7] = 0;
  i32x4 z4 = {0, 0, 0, 0};                                   // groups 2/3: 2D
  i32x8 z8 = {0, 0, 0, 0, 0, 0, 0, 0};
  __builtin_amdgcn_tensor_load_to_lds(g0, g1, z4, z4, z8, 0);
}

// ============================================================
// Tiled GEMM: C[M,N] = A[M,K] * W[N,K]^T   (bf16 WMMA, f32 acc)
// Block tile 64(M) x 128(N), 8 waves, 2x2 WMMA tiles each.
// B stored k-major in LDS so B-fragments are contiguous reads.
// ============================================================
template <typename AT, typename OT>
__global__ __launch_bounds__(256)
void gemm_bf16_wmma(const AT* __restrict__ A, const float* __restrict__ W,
                    OT* __restrict__ Cout, int K, int N) {
  __shared__ bf16 As[64][36];     // 72B rows
  __shared__ bf16 Bst[32][132];   // k-major, 264B rows (66 dw, gcd(33,32)~ok)
  const int tid  = threadIdx.x;
  const int lane = tid & 31, wave = tid >> 5;
  const int hi = lane >> 4, lo = lane & 15;
  const int m0 = blockIdx.y * 64, n0 = blockIdx.x * 128;
  const int wm = (wave & 1) * 32, wn = (wave >> 1) * 32;

  v8f acc[2][2] = {};
  for (int k0 = 0; k0 < K; k0 += 32) {
    { // stage A tile (64 x 32), convert to bf16
      int row = tid >> 2, col = (tid & 3) * 8;
      #pragma unroll
      for (int j = 0; j < 8; ++j)
        As[row][col + j] = ld_bf(&A[(size_t)(m0 + row) * K + k0 + col + j]);
    }
    { // stage B tile = W rows (128 n x 32 k), transposed into LDS
      int row = tid >> 1, col = (tid & 1) * 16;   // row=n, col=k
      #pragma unroll
      for (int j = 0; j < 16; ++j)
        Bst[col + j][row] = f2bf(W[(size_t)(n0 + row) * K + k0 + col + j]);
      if (k0 + 32 < K)  // CDNA5 global_prefetch_b8 path
        __builtin_prefetch(&W[(size_t)(n0 + row) * K + k0 + 32 + col], 0, 1);
    }
    __syncthreads();

    v16bf af[2], bfr[2];
    #pragma unroll
    for (int i = 0; i < 2; ++i) {
      int r = wm + i * 16 + lo, kh = hi * 8;
      af[i] = ld16bf_2x8(&As[r][kh], &As[r][kh + 16]);
    }
    #pragma unroll
    for (int i = 0; i < 2; ++i)
      #pragma unroll
      for (int s = 0; s < 16; ++s)
        bfr[i][s] = Bst[hi * 8 + s + (s < 8 ? 0 : 8)][wn + i * 16 + lo];  // lane = n, elems = k
    #pragma unroll
    for (int i = 0; i < 2; ++i)
      #pragma unroll
      for (int j = 0; j < 2; ++j)
        acc[i][j] = __builtin_amdgcn_wmma_f32_16x16x32_bf16(
            false, af[i], false, bfr[j], (short)0, acc[i][j], false, false);
    __syncthreads();
  }
  #pragma unroll
  for (int i = 0; i < 2; ++i)
    #pragma unroll
    for (int j = 0; j < 2; ++j)
      #pragma unroll
      for (int r = 0; r < 8; ++r) {
        int grow = m0 + wm + i * 16 + r + 8 * hi;
        int gcol = n0 + wn + j * 16 + lo;
        st_out(&Cout[(size_t)grow * N + gcol], acc[i][j][r]);
      }
}

// ============================================================
// RoPE + split qkv(bf16,[tok][3][H][D]) -> Q[bh][s][d],
// Kt[bh][d][s] (transposed for score B-fragments), V[bh][s][d]
// ============================================================
__global__ __launch_bounds__(256)
void rope_split(const bf16* __restrict__ qkv, bf16* __restrict__ Q,
                bf16* __restrict__ Kt, bf16* __restrict__ V) {
  const int token = blockIdx.x;
  const int b = token >> 10, s = token & (SEQ - 1);
  const int tid = threadIdx.x;
  const size_t base = (size_t)token * QKV_N;
  #pragma unroll
  for (int it = 0; it < 2; ++it) {
    int p = it * 256 + tid;            // 0..511 = 16 heads * 32 pairs
    int h = p >> 5, d = p & 31;
    // inv_freq = 10000^(-d/32) = exp2(-(d/32)*log2(10000))
    float ang = (float)s * exp2f(-(float)d * (13.287712379549449f / 32.0f));
    float sn, cs;
    __sincosf(ang, &sn, &cs);
    {
      size_t dst = (((size_t)(b * HEADS + h)) * SEQ + s) * HEAD;
      float x1 = bf2f(qkv[base + 0 * HIDDEN + h * HEAD + d]);
      float x2 = bf2f(qkv[base + 0 * HIDDEN + h * HEAD + d + 32]);
      Q[dst + d]      = f2bf(x1 * cs - x2 * sn);
      Q[dst + d + 32] = f2bf(x2 * cs + x1 * sn);
    }
    {
      size_t dstT = (((size_t)(b * HEADS + h)) * HEAD + d) * SEQ + s;
      float x1 = bf2f(qkv[base + 1 * HIDDEN + h * HEAD + d]);
      float x2 = bf2f(qkv[base + 1 * HIDDEN + h * HEAD + d + 32]);
      Kt[dstT]            = f2bf(x1 * cs - x2 * sn);
      Kt[dstT + 32 * SEQ] = f2bf(x2 * cs + x1 * sn);
    }
  }
  #pragma unroll
  for (int it = 0; it < 4; ++it) {
    int p = it * 256 + tid;            // 0..1023
    int h = p >> 6, d = p & 63;
    size_t dst = (((size_t)(b * HEADS + h)) * SEQ + s) * HEAD;
    V[dst + d] = qkv[base + 2 * HIDDEN + h * HEAD + d];
  }
}

// ============================================================
// Sliding-window attention. Block = 4 waves = 64 query rows.
// TDM stages Kt (64 d-rows x 256 key-cols) and V (208 rows x
// 64) into padded LDS; K/V B-fragment elements are gathered
// one per LDS row (lanes on adjacent cols), conflict-free stride.
// ============================================================
__global__ __launch_bounds__(128)
void swin_attn(const bf16* __restrict__ Q, const bf16* __restrict__ Kt,
               const bf16* __restrict__ V, const int* __restrict__ amask,
               bf16* __restrict__ Out) {
  __shared__ bf16 Kts[64][KCOL];     // 33,024 B
  __shared__ bf16 Vs[KTILE][VROW];   // 27,456 B
  __shared__ bf16 Pl[4][16][34];     //  4,352 B  (total 64,832 <= 64KB)
  const int tid = threadIdx.x, lane = tid & 31, wave = tid >> 5;
  const int hi = lane >> 4, lo = lane & 15;
  const int bh = blockIdx.y, b = bh >> 4, h = bh & 15;
  const int B0 = blockIdx.x * 64;
  const int q0 = B0 + wave * 16;
  const int low = B0 - WIN;          // global key index of LDS tile col/row 0
  const size_t hbase = (size_t)bh * SEQ * HEAD;

  // ---- TDM staging (wave 0 issues; EXEC ignored by tensor ops) ----
  const int start = low < 0 ? 0 : low;   // V front clamp
  const int shift = start - low;         // 0, or 64 when B0==0
  if (wave == 0) {
    // Kt tile: 64 rows (d) x 256 cols (keys [low, low+256)); row stride SEQ.
    // low<0 reads (finite) bytes before Kt -> columns are mask-discarded.
    tdm_load_2d((unsigned)(size_t)&Kts[0][0],
                Kt + (long long)bh * HEAD * SEQ + (long long)low,
                /*tile_d0=*/256, /*tile_d1=*/64,
                /*tdim0=*/SEQ - low, /*tdim1=*/64, /*stride=*/SEQ,
                /*padi=*/6, /*pada=*/0);         // 128dw rows + 1dw pad = 258
    // V tile: (KTILE-shift) rows x 64; rows >= SEQ zero-filled by OOB.
    tdm_load_2d((unsigned)(size_t)&Vs[0][0] + (unsigned)shift * (VROW * 2),
                V + hbase + (size_t)start * HEAD,
                /*tile_d0=*/64, /*tile_d1=*/KTILE - shift,
                /*tdim0=*/64, /*tdim1=*/SEQ - start, /*stride=*/64,
                /*padi=*/4, /*pada=*/0);         // 32dw rows + 1dw pad = 66
    __builtin_amdgcn_s_wait_tensorcnt(0);
  }
  if (shift != 0 && tid < 64) {        // V rows below seq start: make finite 0
    bf16 z = f2bf(0.0f);
    for (int j = 0; j < VROW; ++j) Vs[tid][j] = z;
  }
  __syncthreads();

  // ---- Q A-fragments (K=64 -> two k-steps), vectorized global reads ----
  v16bf qa[2];
  {
    const bf16* rb = Q + hbase + (size_t)(q0 + lo) * HEAD + hi * 8;
    qa[0] = ld16bf_2x8(rb, rb + 16);
    qa[1] = ld16bf_2x8(rb + 32, rb + 48);
  }

  float mrow[8], lrow[8];
  #pragma unroll
  for (int r = 0; r < 8; ++r) { mrow[r] = -3.0e38f; lrow[r] = 0.0f; }
  v8f o[4] = {};

  for (int c = 0; c < 5; ++c) {
    const int cbase = q0 - WIN + c * 32;
    // ---- scores: two 16-wide key tiles, 2 head-dim k-steps each ----
    v8f sc[2] = {};
    #pragma unroll
    for (int t = 0; t < 2; ++t) {
      int lc0 = cbase + t * 16 - low;       // LDS key col, in [0, 256)
      #pragma unroll
      for (int ds = 0; ds < 2; ++ds) {
        v16bf kb;  // lane=key, elems=d
        #pragma unroll
        for (int s = 0; s < 16; ++s)
          kb[s] = Kts[ds * 32 + hi * 8 + s + (s < 8 ? 0 : 8)][lc0 + lo];
        sc[t] = __builtin_amdgcn_wmma_f32_16x16x32_bf16(
            false, qa[ds], false, kb, (short)0, sc[t], false, false);
      }
    }
    // ---- key-column validity (independent of query row) ----
    int  jv[2];
    bool jok[2];
    #pragma unroll
    for (int t = 0; t < 2; ++t) {
      int j = cbase + t * 16 + lo;
      int jc = j < 0 ? 0 : (j >= SEQ ? SEQ - 1 : j);
      jv[t] = j;
      jok[t] = (j >= 0) && (j < SEQ) && (amask[b * SEQ + jc] != 0);
    }
    // ---- mask + scale + online softmax (rows live across 16 lanes) ----
    #pragma unroll
    for (int r = 0; r < 8; ++r) {
      int i = q0 + r + 8 * hi;
      float sv[2];
      #pragma unroll
      for (int t = 0; t < 2; ++t) {
        bool ok = jok[t] && (jv[t] >= i - WIN) && (jv[t] <= i + WIN);
        sv[t] = ok ? sc[t][r] * 0.125f : -3.0e38f;
      }
      float vmax = fmaxf(sv[0], sv[1]);
      #pragma unroll
      for (int d = 8; d >= 1; d >>= 1) vmax = fmaxf(vmax, __shfl_xor(vmax, d, 32));
      float newm = fmaxf(mrow[r], vmax);
      float p0 = (sv[0] < -1.0e37f) ? 0.0f : __expf(sv[0] - newm);
      float p1 = (sv[1] < -1.0e37f) ? 0.0f : __expf(sv[1] - newm);
      float rs = p0 + p1;
      #pragma unroll
      for (int d = 8; d >= 1; d >>= 1) rs += __shfl_xor(rs, d, 32);
      float corr = __expf(mrow[r] - newm);
      lrow[r] = lrow[r] * corr + rs;
      mrow[r] = newm;
      #pragma unroll
      for (int t4 = 0; t4 < 4; ++t4) o[t4][r] *= corr;
      Pl[wave][r + 8 * hi][lo]      = f2bf(p0);
      Pl[wave][r + 8 * hi][lo + 16] = f2bf(p1);
    }
    __syncthreads();
    // ---- P as A-fragment, V as B-fragments: 4 WMMA (64 head dims) ----
    v16bf pa = ld16bf_2x8(&Pl[wave][lo][hi * 8], &Pl[wave][lo][hi * 8 + 16]);
    int vr = cbase - low;                   // LDS V row base, in [0, KTILE)
    #pragma unroll
    for (int t4 = 0; t4 < 4; ++t4) {
      v16bf vb;  // lane=d, elems=key
      #pragma unroll
      for (int s = 0; s < 16; ++s)
        vb[s] = Vs[vr + hi * 8 + s + (s < 8 ? 0 : 8)][t4 * 16 + lo];
      o[t4] = __builtin_amdgcn_wmma_f32_16x16x32_bf16(
          false, pa, false, vb, (short)0, o[t4], false, false);
    }
    __syncthreads();
  }
  // ---- epilogue: normalize, write [tok][h*64+d] bf16 ----
  #pragma unroll
  for (int r = 0; r < 8; ++r) {
    int srow = q0 + r + 8 * hi;
    float inv = lrow[r] > 0.0f ? 1.0f / lrow[r] : 0.0f;
    size_t tok = (size_t)b * SEQ + srow;
    #pragma unroll
    for (int t4 = 0; t4 < 4; ++t4)
      Out[tok * HIDDEN + h * HEAD + t4 * 16 + lo] = f2bf(o[t4][r] * inv);
  }
}

// ============================================================
// Launch
// ============================================================
extern "C" void kernel_launch(void* const* d_in, const int* in_sizes, int n_in,
                              void* d_out, int out_size, void* d_ws, size_t ws_size,
                              hipStream_t stream) {
  (void)in_sizes; (void)n_in; (void)out_size; (void)ws_size;
  const float* hidden = (const float*)d_in[0];
  // d_in[1]=cu_seqlens, d_in[2]=max_seqlen, d_in[3]=indices : fixed-shape case
  const int*   amask  = (const int*)d_in[4];
  const float* Wqkv   = (const float*)d_in[5];
  const float* Wo     = (const float*)d_in[6];

  char* ws = (char*)d_ws;
  bf16* qkv_bf = (bf16*)(ws + 0);          // 8192*3072*2  = 50331648
  bf16* Q      = (bf16*)(ws + 50331648);   // 16777216
  bf16* V      = (bf16*)(ws + 67108864);   // 16777216 (precedes Kt: see TDM low<0)
  bf16* Kt     = (bf16*)(ws + 83886080);   // 16777216
  bf16* attn   = (bf16*)(ws + 100663296);  // 16777216
  float* out   = (float*)d_out;

  // 1) qkv = hidden @ Wqkv^T   (8192x1024 * 1024x3072)
  gemm_bf16_wmma<float, bf16>
      <<<dim3(QKV_N / 128, NTOK / 64), 256, 0, stream>>>(hidden, Wqkv, qkv_bf,
                                                         HIDDEN, QKV_N);
  // 2) RoPE + split into per-head layouts (K transposed)
  rope_split<<<NTOK, 256, 0, stream>>>(qkv_bf, Q, Kt, V);
  // 3) sliding-window attention (TDM-staged Kt/V)
  swin_attn<<<dim3(SEQ / 64, BATCH * HEADS), 128, 0, stream>>>(Q, Kt, V, amask,
                                                               attn);
  // 4) out = attn @ Wo^T
  gemm_bf16_wmma<bf16, float>
      <<<dim3(HIDDEN / 128, NTOK / 64), 256, 0, stream>>>(attn, Wo, out,
                                                          HIDDEN, HIDDEN);
}
